// Head_17549236372243
// MI455X (gfx1250) — matrix operand order repaired and checked by reading, not compiled, not run on hardware
//
#include <hip/hip_runtime.h>
#include <math.h>

typedef __attribute__((ext_vector_type(16))) _Float16 v16h;
typedef __attribute__((ext_vector_type(8)))  float    v8f;

#define Bsz 16
#define Cch 64
#define Hh  256
#define Ww  256
#define HW  (Hh*Ww)
#define PIX 128           // pixels (threads) per block
#define S16 33            // dword stride per pixel, f16 activation buffer (padded)
#define S32 66            // float stride per pixel, f32 mid buffer (padded)
#define ROWS_PER_WG 4

__device__ __forceinline__ float gelu_exact(float x) {
    return 0.5f * x * (1.0f + erff(x * 0.70710678118654752f));
}

__device__ __forceinline__ unsigned pack_f16x2(float a, float b) {
    unsigned short ua = __builtin_bit_cast(unsigned short, (_Float16)a);
    unsigned short ub = __builtin_bit_cast(unsigned short, (_Float16)b);
    return (unsigned)ua | ((unsigned)ub << 16);
}

__global__ __launch_bounds__(PIX) void convnext_fused_kernel(
    const float* __restrict__ in,     const float* __restrict__ conv_w,
    const float* __restrict__ conv_b, const float* __restrict__ pw1_w,
    const float* __restrict__ pw1_b,  const float* __restrict__ pw2_w,
    const float* __restrict__ pw2_b,  const float* __restrict__ norm_g,
    const float* __restrict__ norm_b, float* __restrict__ out)
{
    __shared__ float    s_convw[576];
    __shared__ float    s_convb[Cch], s_g[Cch], s_b[Cch], s_pw1b[Cch];
    __shared__ float    s_pw2w[3*Cch];
    __shared__ float    s_pw2b[4];
    __shared__ unsigned s_act16[PIX * S16];   // post-GELU activations, packed f16 pairs
    __shared__ float    s_mid[PIX * S32];     // conv raw / pw1 output (f32)

    const int t     = threadIdx.x;
    const int lane  = t & 31;
    const int wave  = t >> 5;
    const int lhalf = (lane >> 4) & 1;        // 0: lanes 0-15, 1: lanes 16-31
    const int lmod  = lane & 15;

    // ---- cooperative weight staging ----
    for (int i = t; i < 576; i += PIX) s_convw[i] = conv_w[i];
    if (t < Cch) { s_convb[t]=conv_b[t]; s_g[t]=norm_g[t]; s_b[t]=norm_b[t]; s_pw1b[t]=pw1_b[t]; }
    for (int i = t; i < 3*Cch; i += PIX) s_pw2w[i] = pw2_w[i];
    if (t < 3) s_pw2b[t] = pw2_b[t];
    if (t == 3) s_pw2b[3] = 0.f;

    // ---- preload pw1 weights as persistent WMMA B-fragments (f16) ----
    // D[p][o] = sum_c X[p][c]*W1[o][c]  =>  B[k][n] = W1[n][k]
    // 32x16 f16 B, k order used here (contiguous, same as A): VGPR v, lanes0-15: K=2v,2v+1; lanes16-31: K=16+2v,17+2v; N=lane%16
    v16h bfrag[4][2];
    #pragma unroll
    for (int nt = 0; nt < 4; ++nt) {
        const int N = nt*16 + lmod;
        #pragma unroll
        for (int kk = 0; kk < 2; ++kk) {
            const float* pw = pw1_w + N*Cch + kk*32 + lhalf*16;
            union { unsigned u[8]; v16h h; } Bf;
            #pragma unroll
            for (int v = 0; v < 8; ++v) Bf.u[v] = pack_f16x2(pw[2*v], pw[2*v+1]);
            bfrag[nt][kk] = Bf.h;
        }
    }
    __syncthreads();

    const int wg     = blockIdx.x;
    const int halfw  = wg & 1;                 // which half of the W=256 row
    const int rowgrp = wg >> 1;
    const int wpix   = halfw*PIX + t;          // global w coordinate for this thread

    for (int it = 0; it < ROWS_PER_WG; ++it) {
        const int r  = rowgrp*ROWS_PER_WG + it;   // 0..4095
        const int bb = r >> 8;
        const int hh = r & 255;
        const float* inb = in + (size_t)bb * Cch * HW;

        // ======== Phase A: depthwise 3x3 conv + LN1 stats ========
        float sum = 0.f, sumsq = 0.f;
        const bool wm_ok = (wpix - 1) >= 0;
        const bool wp_ok = (wpix + 1) < Ww;
        for (int c = 0; c < Cch; ++c) {
            const float* cw = &s_convw[c*9];
            const float* pc = inb + (size_t)c * HW;
            float acc = s_convb[c];
            #pragma unroll
            for (int dh = -1; dh <= 1; ++dh) {
                const int y = hh + dh;                     // uniform across block
                if (y >= 0 && y < Hh) {
                    const float* pr = pc + y*Ww;
                    const float* w3 = cw + (dh+1)*3;
                    if (wm_ok) acc += pr[wpix-1]*w3[0];
                    acc += pr[wpix]*w3[1];
                    if (wp_ok) acc += pr[wpix+1]*w3[2];
                }
            }
            s_mid[t*S32 + c] = acc;
            sum += acc; sumsq += acc*acc;
        }
        {   // LN1 + GELU, pack to f16 activation buffer
            const float mu = sum * (1.0f/64.0f);
            float var = fmaxf(sumsq * (1.0f/64.0f) - mu*mu, 0.0f);
            const float rs = rsqrtf(var + 1e-5f);
            for (int k = 0; k < 32; ++k) {
                const float x0 = (s_mid[t*S32 + 2*k]   - mu)*rs*s_g[2*k]   + s_b[2*k];
                const float x1 = (s_mid[t*S32 + 2*k+1] - mu)*rs*s_g[2*k+1] + s_b[2*k+1];
                s_act16[t*S16 + k] = pack_f16x2(gelu_exact(x0), gelu_exact(x1));
            }
        }
        __syncthreads();

        // ======== Phase B: pw1 64x64 GEMM via v_wmma_f32_16x16x32_f16 ========
        // A k order used here (16x32 f16, contiguous, same as B): lanes0-15 M=0-15, VGPR v -> K dword v;
        //                       lanes16-31 same M, +8 dwords; K-step adds 16 dwords.
        #pragma unroll
        for (int mt = 0; mt < 2; ++mt) {
            const int Mbase = wave*32 + mt*16;
            const unsigned* pa = &s_act16[(Mbase + lmod) * S16];
            union { unsigned u[8]; v16h h; } A0, A1;
            #pragma unroll
            for (int v = 0; v < 8; ++v) {
                const int kdw = v + lhalf*8;
                A0.u[v] = pa[kdw];
                A1.u[v] = pa[16 + kdw];
            }
            #pragma unroll
            for (int nt = 0; nt < 4; ++nt) {
                v8f acc = {};
                acc = __builtin_amdgcn_wmma_f32_16x16x32_f16(
                        false, A0.h, false, bfrag[nt][0], (short)0, acc, false, false);
                acc = __builtin_amdgcn_wmma_f32_16x16x32_f16(
                        false, A1.h, false, bfrag[nt][1], (short)0, acc, false, false);
                const int   N    = nt*16 + lmod;
                const float bias = s_pw1b[N];
                const int   pixb = Mbase + lhalf*8;   // D: VGPR v -> M = v (+8 for hi lanes)
                #pragma unroll
                for (int v = 0; v < 8; ++v)
                    s_mid[(pixb + v)*S32 + N] = acc[v] + bias;
            }
        }
        __syncthreads();

        // ======== Phase C: LN2 + GELU + pw2 (64->3) + coalesced NCHW store ========
        {
            float sum2 = 0.f, ssq2 = 0.f;
            for (int c = 0; c < Cch; ++c) {
                const float v = s_mid[t*S32 + c];
                sum2 += v; ssq2 += v*v;
            }
            const float mu = sum2 * (1.0f/64.0f);
            float var = fmaxf(ssq2 * (1.0f/64.0f) - mu*mu, 0.0f);
            const float rs = rsqrtf(var + 1e-5f);
            float o0 = s_pw2b[0], o1 = s_pw2b[1], o2 = s_pw2b[2];
            for (int c = 0; c < Cch; ++c) {
                const float y = gelu_exact((s_mid[t*S32 + c] - mu)*rs*s_g[c] + s_b[c]);
                o0 += y * s_pw2w[c];
                o1 += y * s_pw2w[64 + c];
                o2 += y * s_pw2w[128 + c];
            }
            float* ob = out + (size_t)bb*3*HW + hh*Ww + wpix;
            ob[0*HW] = o0;
            ob[1*HW] = o1;
            ob[2*HW] = o2;
        }
        __syncthreads();   // buffers reused next row
    }
}

extern "C" void kernel_launch(void* const* d_in, const int* in_sizes, int n_in,
                              void* d_out, int out_size, void* d_ws, size_t ws_size,
                              hipStream_t stream) {
    const float* in     = (const float*)d_in[0];
    const float* conv_w = (const float*)d_in[1];
    const float* conv_b = (const float*)d_in[2];
    const float* pw1_w  = (const float*)d_in[3];
    const float* pw1_b  = (const float*)d_in[4];
    const float* pw2_w  = (const float*)d_in[5];
    const float* pw2_b  = (const float*)d_in[6];
    const float* norm_g = (const float*)d_in[7];
    const float* norm_b = (const float*)d_in[8];
    float* outp = (float*)d_out;

    const int grid = (Bsz*Hh / ROWS_PER_WG) * (Ww / PIX);   // 1024 * 2 = 2048 blocks
    convnext_fused_kernel<<<grid, PIX, 0, stream>>>(
        in, conv_w, conv_b, pw1_w, pw1_b, pw2_w, pw2_b, norm_g, norm_b, outp);
}
